// MultiheadCrossAttention_37675453120634
// MI455X (gfx1250) — compile-verified, matrix operand order checked by reading, not run on hardware
//
#include <hip/hip_runtime.h>
#include <hip/hip_bf16.h>

typedef __bf16 bf16;
typedef __attribute__((ext_vector_type(16))) __bf16 bf16x16;
typedef __attribute__((ext_vector_type(8)))  __bf16 bf16x8;
typedef __attribute__((ext_vector_type(8)))  float   v8f;
typedef __attribute__((ext_vector_type(4)))  int     v4i;

#ifndef __has_builtin
#define __has_builtin(x) 0
#endif

#if __has_builtin(__builtin_amdgcn_global_load_async_to_lds_b128)
#define HAVE_ASYNC_LDS 1
#endif
#if __has_builtin(__builtin_amdgcn_s_wait_asynccnt)
#define HAVE_WAIT_ASYNC 1
#endif

typedef __attribute__((address_space(1))) v4i* as1_v4i;
typedef __attribute__((address_space(3))) v4i* as3_v4i;

// 16-byte global -> LDS copy. Uses CDNA5 async-to-LDS (ASYNCcnt) when the
// toolchain exposes the builtin; falls back to a VGPR-staged copy otherwise.
__device__ __forceinline__ void copy16_g2l(const bf16* g, bf16* l) {
#if defined(HAVE_ASYNC_LDS)
    v4i* gp = (v4i*)(void*)const_cast<bf16*>(g);
    v4i* lp = (v4i*)(void*)l;
    __builtin_amdgcn_global_load_async_to_lds_b128((as1_v4i)gp, (as3_v4i)lp, 0, 0);
#else
    *(bf16x8*)l = *(const bf16x8*)g;
#endif
}

__device__ __forceinline__ void async_copies_wait() {
#if defined(HAVE_ASYNC_LDS)
#if defined(HAVE_WAIT_ASYNC)
    __builtin_amdgcn_s_wait_asynccnt(0);
#else
    asm volatile("s_wait_asynccnt 0x0" ::: "memory");
#endif
#endif
}

// ---------------------------------------------------------------------------
// CDNA5 16-bit A/B operand fragment loader.
// Layout (ISA 7.12.2, 16-bit A 16x32): lane l holds row m=l&15;
//   lanes 0-15:  K = {0..7, 16..23}; lanes 16-31: K = {8..15, 24..31}.
// `row` points at element [m][0] of a 32-wide K strip (row-major, any stride).
// ---------------------------------------------------------------------------
__device__ __forceinline__ bf16x16 load_frag_row(const bf16* row) {
    const int lane = threadIdx.x & 31;
    const int kb   = (lane >> 4) * 8;           // 0 or 8
    bf16x8 lo = *(const bf16x8*)(row + kb);
    bf16x8 hi = *(const bf16x8*)(row + 16 + kb);
    bf16x16 f;
#pragma unroll
    for (int j = 0; j < 8; ++j) { f[j] = lo[j]; f[8 + j] = hi[j]; }
    return f;
}

// ---------------------------------------------------------------------------
// f32 -> bf16 conversion (grid-stride)
// ---------------------------------------------------------------------------
__global__ void f32_to_bf16_kernel(const float* __restrict__ in,
                                   bf16* __restrict__ out, int n) {
    int i = blockIdx.x * blockDim.x + threadIdx.x;
    int stride = gridDim.x * blockDim.x;
    for (; i < n; i += stride) out[i] = (bf16)in[i];
}

// ---------------------------------------------------------------------------
// Tiled bf16 WMMA GEMM: C[M,N] = A[M,K] * B[K,N] + bias[N]
// BM=128, BN=64, BK=32; 256 threads = 8 waves; wave w -> rows [16w,16w+16).
// ---------------------------------------------------------------------------
#define GBM 128
#define GBN 64
#define GBK 32

__global__ __launch_bounds__(256) void gemm_bf16_wmma(
    const bf16* __restrict__ A, const bf16* __restrict__ B,
    const float* __restrict__ bias, void* __restrict__ Cout,
    int M, int N, int K, int out_f32)
{
    __shared__ __align__(16) bf16 sA [GBM][GBK + 8];   // row stride 80B (16B mult)
    __shared__ __align__(16) bf16 sBt[GBN][GBK + 8];   // B transposed: [n][k]

    const int tid  = threadIdx.x;
    const int lane = tid & 31;
    const int w    = tid >> 5;                  // 0..7
    const int row0 = blockIdx.y * GBM;
    const int col0 = blockIdx.x * GBN;

    v8f acc[4] = {};                            // 4 x (16x16) tiles, cols nv*16

    for (int k0 = 0; k0 < K; k0 += GBK) {
        __syncthreads();
        // A tile 128x32: 512 chunks of 8 bf16; async global->LDS, 2 per thread
#pragma unroll
        for (int it = 0; it < 2; ++it) {
            int idx = tid + it * 256;           // 0..511
            int r = idx >> 2;                   // 0..127
            int c = (idx & 3) * 8;              // 0,8,16,24
            copy16_g2l(A + (size_t)(row0 + r) * K + k0 + c, &sA[r][c]);
        }
        // B tile 32x64: 256 chunks of 8, transposed scatter into sBt
        {
            int r = tid >> 3;                   // 0..31 (k)
            int c = (tid & 7) * 8;              // 0..56 (n)
            bf16x8 bv = *(const bf16x8*)(B + (size_t)(k0 + r) * N + col0 + c);
#pragma unroll
            for (int j = 0; j < 8; ++j) sBt[c + j][r] = bv[j];
        }
        async_copies_wait();
        __syncthreads();

        if (k0 + GBK < K) {  // prefetch next A tile row -> global_prefetch_b8
            __builtin_prefetch(A + (size_t)(row0 + (tid >> 1)) * K + k0 + GBK, 0, 1);
        }

        const int m = w * 16 + (lane & 15);
        bf16x16 afrag = load_frag_row(&sA[m][0]);
#pragma unroll
        for (int nv = 0; nv < 4; ++nv) {
            const int n = nv * 16 + (lane & 15);
            bf16x16 bfrag = load_frag_row(&sBt[n][0]);
            acc[nv] = __builtin_amdgcn_wmma_f32_16x16x32_bf16(
                false, afrag, false, bfrag, (short)0, acc[nv], false, false);
        }
    }

    // Epilogue: C layout (ISA 7.12.2): lane l, vgpr i -> row = i + 8*(l>>4),
    // col = l&15 within the 16x16 tile.
    const int hi2  = lane >> 4;
    const int nin  = lane & 15;
#pragma unroll
    for (int nv = 0; nv < 4; ++nv) {
        int ncol = col0 + nv * 16 + nin;
        float bval = bias ? bias[ncol] : 0.f;
#pragma unroll
        for (int i = 0; i < 8; ++i) {
            int r = row0 + w * 16 + hi2 * 8 + i;
            float val = acc[nv][i] + bval;
            if (out_f32) ((float*)Cout)[(size_t)r * N + ncol] = val;
            else         ((bf16*)Cout)[(size_t)r * N + ncol] = (bf16)val;
        }
    }
}

// ---------------------------------------------------------------------------
// Flash attention, bf16 WMMA, f32 accumulate.
// qk: [N][2D] (q = cols 0..D-1, k = cols D..2D-1), v: [N][D], out bf16 [N][D].
// Grid: (N/64, H). Block: 128 threads = 4 waves; wave w -> 16 query rows.
// ---------------------------------------------------------------------------
#define ABQ 64
#define ABK 32

__global__ __launch_bounds__(128) void attn_flash_wmma(
    const bf16* __restrict__ qk, const bf16* __restrict__ v,
    bf16* __restrict__ out, int N, int D, int nheads)
{
    const int dk   = D / nheads;                // 64
    const int h    = blockIdx.y;
    const int q0   = blockIdx.x * ABQ;
    const int tid  = threadIdx.x;
    const int lane = tid & 31;
    const int w    = tid >> 5;                  // 0..3
    const int twoD = 2 * D;

    __shared__ __align__(16) bf16 sK [ABK][64 + 8];       // [krow][d]
    __shared__ __align__(16) bf16 sVt[64][ABK + 8];       // [d][krow]
    __shared__ __align__(16) bf16 sP [4][16][ABK + 8];    // per-wave P staging

    // Q fragments: wave's 16 rows x 64 dims, resident in registers.
    const int mrow = q0 + w * 16 + (lane & 15);
    const bf16* qrow = qk + (size_t)mrow * twoD + h * dk;
    bf16x16 qfrag[2];
    qfrag[0] = load_frag_row(qrow);
    qfrag[1] = load_frag_row(qrow + 32);

    v8f acc[4] = {};
    float run_m[8], run_l[8];
#pragma unroll
    for (int i = 0; i < 8; ++i) { run_m[i] = -3.0e38f; run_l[i] = 0.f; }

    const float scale = 0.125f;                 // 1/sqrt(64)

    for (int kb0 = 0; kb0 < N; kb0 += ABK) {
        __syncthreads();
        // K tile 32x64 (row-major as in memory): async global->LDS
#pragma unroll
        for (int it = 0; it < 2; ++it) {
            int idx = tid + it * 128;           // 0..255
            int r = idx >> 3;                   // 0..31
            int c = (idx & 7) * 8;
            copy16_g2l(qk + (size_t)(kb0 + r) * twoD + D + h * dk + c, &sK[r][c]);
        }
        // V tile 32x64, stored transposed [d][krow]
#pragma unroll
        for (int it = 0; it < 2; ++it) {
            int idx = tid + it * 128;
            int r = idx >> 3;
            int c = (idx & 7) * 8;
            bf16x8 vv = *(const bf16x8*)(v + (size_t)(kb0 + r) * D + h * dk + c);
#pragma unroll
            for (int j = 0; j < 8; ++j) sVt[c + j][r] = vv[j];
        }
        async_copies_wait();
        __syncthreads();

        // S = Q * K^T : two 16x16 tiles (key cols 0..15, 16..31), K-dim 64 = 2 chunks
        v8f s[2] = {};
#pragma unroll
        for (int nh = 0; nh < 2; ++nh) {
            const int n = nh * 16 + (lane & 15);
#pragma unroll
            for (int c = 0; c < 2; ++c) {
                bf16x16 kf = load_frag_row(&sK[n][c * 32]);
                s[nh] = __builtin_amdgcn_wmma_f32_16x16x32_bf16(
                    false, qfrag[c], false, kf, (short)0, s[nh], false, false);
            }
        }

        // Online softmax: row m = i + 8*(lane>>4) lives across the 16 lanes of
        // one half-wave -> xor-shuffle reduction with masks 8..1 stays in-half.
        float p0[8], p1[8], corr[8];
#pragma unroll
        for (int i = 0; i < 8; ++i) {
            float s0 = s[0][i] * scale;
            float s1 = s[1][i] * scale;
            float mt = fmaxf(s0, s1);
#pragma unroll
            for (int msk = 8; msk >= 1; msk >>= 1)
                mt = fmaxf(mt, __shfl_xor(mt, msk, 32));
            float nm = fmaxf(run_m[i], mt);
            corr[i] = __expf(run_m[i] - nm);
            run_m[i] = nm;
            p0[i] = __expf(s0 - nm);
            p1[i] = __expf(s1 - nm);
            float rs = p0[i] + p1[i];
#pragma unroll
            for (int msk = 8; msk >= 1; msk >>= 1)
                rs += __shfl_xor(rs, msk, 32);
            run_l[i] = run_l[i] * corr[i] + rs;
        }
        // Rescale accumulated O by per-row correction
#pragma unroll
        for (int nv = 0; nv < 4; ++nv)
#pragma unroll
            for (int i = 0; i < 8; ++i)
                acc[nv][i] *= corr[i];

        // Re-layout P (C-layout) -> A-fragment layout via per-wave LDS staging.
        const int hi2 = lane >> 4;
        const int nin = lane & 15;
#pragma unroll
        for (int i = 0; i < 8; ++i) {
            sP[w][hi2 * 8 + i][nin]      = (bf16)p0[i];
            sP[w][hi2 * 8 + i][16 + nin] = (bf16)p1[i];
        }
        // Same-wave LDS ops are in-order; no barrier needed for wave-local sP.
        bf16x16 pfrag = load_frag_row(&sP[w][lane & 15][0]);

        // O += P * V  (K-dim 32 = one WMMA per 16-wide d tile)
#pragma unroll
        for (int nv = 0; nv < 4; ++nv) {
            const int d = nv * 16 + (lane & 15);
            bf16x16 vf = load_frag_row(&sVt[d][0]);
            acc[nv] = __builtin_amdgcn_wmma_f32_16x16x32_bf16(
                false, pfrag, false, vf, (short)0, acc[nv], false, false);
        }
    }

    // Normalize and write bf16 [N][D] (head h occupies cols h*dk .. h*dk+63)
    const int hi2 = lane >> 4;
    const int nin = lane & 15;
#pragma unroll
    for (int nv = 0; nv < 4; ++nv) {
#pragma unroll
        for (int i = 0; i < 8; ++i) {
            int r = q0 + w * 16 + hi2 * 8 + i;
            float val = acc[nv][i] / run_l[i];
            out[(size_t)r * D + h * dk + nv * 16 + nin] = (bf16)val;
        }
    }
}

// ---------------------------------------------------------------------------
// Launch: convert -> QK/V projections -> flash attention -> O projection
// ---------------------------------------------------------------------------
extern "C" void kernel_launch(void* const* d_in, const int* in_sizes, int n_in,
                              void* d_out, int out_size, void* d_ws, size_t ws_size,
                              hipStream_t stream) {
    const int N = 4096, D = 512, H = 8;
    const float* hs      = (const float*)d_in[0];
    const float* hd      = (const float*)d_in[1];
    const float* qk_kern = (const float*)d_in[2];
    const float* qk_bias = (const float*)d_in[3];
    const float* v_kern  = (const float*)d_in[4];
    const float* v_bias  = (const float*)d_in[5];
    const float* o_kern  = (const float*)d_in[6];
    const float* o_bias  = (const float*)d_in[7];

    char* ws = (char*)d_ws;
    size_t off = 0;
    auto alloc = [&](size_t bytes) {
        char* p = ws + off;
        off += (bytes + 255) & ~size_t(255);
        return p;
    };
    bf16* hs_bf   = (bf16*)alloc((size_t)N * D * 2);
    bf16* hd_bf   = (bf16*)alloc((size_t)N * D * 2);
    bf16* qkk_bf  = (bf16*)alloc((size_t)D * 2 * D * 2);
    bf16* vk_bf   = (bf16*)alloc((size_t)D * D * 2);
    bf16* ok_bf   = (bf16*)alloc((size_t)D * D * 2);
    bf16* qkout   = (bf16*)alloc((size_t)N * 2 * D * 2);
    bf16* vout    = (bf16*)alloc((size_t)N * D * 2);
    bf16* attnout = (bf16*)alloc((size_t)N * D * 2);

    auto cvt = [&](const float* src, bf16* dst, int n) {
        int blocks = (n + 256 * 8 - 1) / (256 * 8);
        if (blocks > 2048) blocks = 2048;
        f32_to_bf16_kernel<<<blocks, 256, 0, stream>>>(src, dst, n);
    };
    cvt(hs,      hs_bf,  N * D);
    cvt(hd,      hd_bf,  N * D);
    cvt(qk_kern, qkk_bf, D * 2 * D);
    cvt(v_kern,  vk_bf,  D * D);
    cvt(o_kern,  ok_bf,  D * D);

    // QK projection: [4096,512] x [512,1024] -> bf16 [4096,1024]
    gemm_bf16_wmma<<<dim3(2 * D / GBN, N / GBM), 256, 0, stream>>>(
        hs_bf, qkk_bf, qk_bias, qkout, N, 2 * D, D, 0);
    // V projection: [4096,512] x [512,512] -> bf16 [4096,512]
    gemm_bf16_wmma<<<dim3(D / GBN, N / GBM), 256, 0, stream>>>(
        hd_bf, vk_bf, v_bias, vout, N, D, D, 0);
    // Flash attention per (query block, head)
    attn_flash_wmma<<<dim3(N / ABQ, H), 128, 0, stream>>>(
        qkout, vout, attnout, N, D, H);
    // Output projection: [4096,512] x [512,512] + bias -> f32 d_out
    gemm_bf16_wmma<<<dim3(D / GBN, N / GBM), 256, 0, stream>>>(
        attnout, ok_bf, o_bias, (float*)d_out, N, D, D, 1);

    (void)in_sizes; (void)n_in; (void)out_size; (void)ws_size;
}
